// EAEF_87101936763049
// MI455X (gfx1250) — compile-verified, matrix operand order checked by reading, not run on hardware
//
#include <hip/hip_runtime.h>
#include <cstdint>

#define B_    8
#define NFULL 16384
#define NP    2048
#define KNN_  32

typedef __attribute__((ext_vector_type(16))) _Float16 v16h;
typedef __attribute__((ext_vector_type(8)))  float    v8f;

// ------------------------------------------------------------------
// Farthest point sampling: one block per batch, dists in registers.
// ------------------------------------------------------------------
__global__ __launch_bounds__(256) void fps_kernel(const float* __restrict__ x,
                                                  float* __restrict__ partial) {
  const int b = blockIdx.x;
  const int t = threadIdx.x;
  const int lane = t & 31, wave = t >> 5;
  __shared__ float s_val[8];
  __shared__ int   s_idx[8];
  __shared__ float s_p[3];
  const float* xb = x + (size_t)b * NFULL * 3;
  const float x0x = xb[0], x0y = xb[1], x0z = xb[2];
  float dist[64];
#pragma unroll 4
  for (int j = 0; j < 64; ++j) {
    int i = j * 256 + t;
    float dx = xb[i*3+0]-x0x, dy = xb[i*3+1]-x0y, dz = xb[i*3+2]-x0z;
    dist[j] = dx*dx + dy*dy + dz*dz;
  }
  if (t == 0) {
    float* p0 = partial + (size_t)b * NP * 3;
    p0[0] = x0x; p0[1] = x0y; p0[2] = x0z;
  }
  for (int s = 1; s < NP; ++s) {
    float bv = -1.0f; int bi = 0x7fffffff;
#pragma unroll 4
    for (int j = 0; j < 64; ++j) {
      int i = j * 256 + t;
      if (dist[j] > bv) { bv = dist[j]; bi = i; }   // ascending i: strict > keeps first index
    }
#pragma unroll
    for (int m = 1; m < 32; m <<= 1) {
      float ov = __shfl_xor(bv, m);
      int   oi = __shfl_xor(bi, m);
      if (ov > bv || (ov == bv && oi < bi)) { bv = ov; bi = oi; }
    }
    if (lane == 0) { s_val[wave] = bv; s_idx[wave] = bi; }
    __syncthreads();
    if (t == 0) {
      float fv = s_val[0]; int fi = s_idx[0];
      for (int wv = 1; wv < 8; ++wv)
        if (s_val[wv] > fv || (s_val[wv] == fv && s_idx[wv] < fi)) { fv = s_val[wv]; fi = s_idx[wv]; }
      float px = xb[fi*3+0], py = xb[fi*3+1], pz = xb[fi*3+2];
      float* pd = partial + ((size_t)b * NP + s) * 3;
      pd[0] = px; pd[1] = py; pd[2] = pz;
      s_p[0] = px; s_p[1] = py; s_p[2] = pz;
    }
    __syncthreads();
    const float px = s_p[0], py = s_p[1], pz = s_p[2];
#pragma unroll 4
    for (int j = 0; j < 64; ++j) {
      int i = j * 256 + t;
      float dx = xb[i*3+0]-px, dy = xb[i*3+1]-py, dz = xb[i*3+2]-pz;
      float dd = dx*dx + dy*dy + dz*dz;
      dist[j] = fminf(dist[j], dd);
    }
  }
}

// ------------------------------------------------------------------
// kNN (top-32 by neg_dist = 2*inner - |q|^2 - |m|^2, tie: lower idx)
// ------------------------------------------------------------------
template <int C, bool COORD>
__global__ __launch_bounds__(128) void knn_kernel(const float* __restrict__ feat,
                                                  int* __restrict__ nbr) {
  const int b = blockIdx.y;
  const int tid = threadIdx.x;
  const int n = blockIdx.x * 128 + tid;
  __shared__ float sf[C][128];
  __shared__ float sxx[128];
  float q[C]; float qq = 0.f;
#pragma unroll
  for (int c = 0; c < C; ++c) {
    float v = COORD ? feat[((size_t)b*NP + n)*3 + c]
                    : feat[((size_t)b*C + c)*NP + n];
    q[c] = v; qq += v * v;
  }
  float bestv[KNN_]; int besti[KNN_];
#pragma unroll
  for (int k = 0; k < KNN_; ++k) { bestv[k] = -3.4e38f; besti[k] = 0x7fffffff; }
  float worstv = -3.4e38f; int worsti = 0x7fffffff;
  for (int m0 = 0; m0 < NP; m0 += 128) {
#pragma unroll
    for (int c = 0; c < C; ++c)
      sf[c][tid] = COORD ? feat[((size_t)b*NP + m0 + tid)*3 + c]
                         : feat[((size_t)b*C + c)*NP + m0 + tid];
    __syncthreads();
    float xx = 0.f;
#pragma unroll
    for (int c = 0; c < C; ++c) xx += sf[c][tid] * sf[c][tid];
    sxx[tid] = xx;
    __syncthreads();
    for (int mm = 0; mm < 128; ++mm) {
      float inner = 0.f;
#pragma unroll
      for (int c = 0; c < C; ++c) inner += q[c] * sf[c][mm];
      float nd = 2.f * inner - qq - sxx[mm];
      int mi = m0 + mm;
      if (nd > worstv || (nd == worstv && mi < worsti)) {
        int pos = KNN_ - 1;
        while (pos > 0 && (nd > bestv[pos-1] || (nd == bestv[pos-1] && mi < besti[pos-1]))) {
          bestv[pos] = bestv[pos-1]; besti[pos] = besti[pos-1]; --pos;
        }
        bestv[pos] = nd; besti[pos] = mi;
        worstv = bestv[KNN_-1]; worsti = besti[KNN_-1];
      }
    }
    __syncthreads();
  }
  int* dst = nbr + ((size_t)b*NP + n)*KNN_;
#pragma unroll
  for (int k = 0; k < KNN_; ++k) dst[k] = besti[k];
}

// ------------------------------------------------------------------
// Weight conversion f32 -> padded f16 (row-major D x C2P)
// ------------------------------------------------------------------
__global__ void cvt_kernel(const float* __restrict__ W, _Float16* __restrict__ W16,
                           int D, int C2, int C2P) {
  int i = blockIdx.x * 256 + threadIdx.x;
  if (i >= D * C2P) return;
  int d = i / C2P, c = i % C2P;
  W16[i] = (c < C2) ? (_Float16)W[d * C2 + c] : (_Float16)0.f;
}

// ------------------------------------------------------------------
// WMMA operand builders: 16-bit A 16x32 layout (ISA 7.12.2):
// lanes 0-15 / 16-31 -> K halves offset by 8; VGPR j<4 -> K=2j, j>=4 -> K=16+2(j-4)
// ------------------------------------------------------------------
__device__ inline v16h load_tile16(const _Float16* __restrict__ base, int rowStride,
                                   int rtile, int kstep, int lane) {
  union { v16h v; uint32_t u[8]; } r;
  const int row = rtile * 16 + (lane & 15);
  const int h = (lane >> 4) & 1;
  const _Float16* p = base + row * rowStride + kstep * 32 + 8 * h;
#pragma unroll
  for (int j = 0; j < 8; ++j) {
    int k = (j < 4) ? (2 * j) : (16 + 2 * (j - 4));
    r.u[j] = *reinterpret_cast<const uint32_t*>(p + k);
  }
  return r.v;
}

// ------------------------------------------------------------------
// Fused gather + 32-neighbor attention. 4 waves/block, 1 point/wave.
// Q = F Wq^T + bq, K = F Wk^T + bk, logits = Q.K/sqrt(D), softmax,
// out = sum_k attn*(F Wv^T) + bv + p   (since sum(attn)=1)
// ------------------------------------------------------------------
template <int C, int C2P, int D, bool COORD>
__global__ __launch_bounds__(128) void attn_kernel(
    const float* __restrict__ feat, const int* __restrict__ nbrIdx,
    const float* __restrict__ partial,
    const _Float16* __restrict__ Wq16, const _Float16* __restrict__ Wk16,
    const _Float16* __restrict__ Wv16,
    const float* __restrict__ bq, const float* __restrict__ bk,
    const float* __restrict__ bv, const float* __restrict__ Wp,
    const float* __restrict__ bp, float* __restrict__ out) {
  const int tid = threadIdx.x;
  const int w = tid >> 5, l = tid & 31;
  const int b = blockIdx.y;
  const int n = blockIdx.x * 4 + w;
  __shared__ __align__(16) _Float16 Fsh[4][32][C2P];
  __shared__ float s_logit[4][32];
  __shared__ float s_attn[4][32];

  if (tid < 32) {  // warm weights into cache (global_prefetch_b8)
    __builtin_prefetch(Wq16 + (size_t)tid * 64, 0, 1);
    __builtin_prefetch(Wk16 + (size_t)tid * 64, 0, 1);
    __builtin_prefetch(Wv16 + (size_t)tid * 64, 0, 1);
  }

  // ---- gather neighbor features: row l = [nbr - ctr, ctr] (f16) ----
  const int myIdx = nbrIdx[((size_t)b * NP + n) * KNN_ + l];
  _Float16* Fr = &Fsh[w][l][0];
  if (COORD) {
    const float* pc = partial + ((size_t)b * NP + n) * 3;
    const float* pb = partial + ((size_t)b * NP + myIdx) * 3;
#pragma unroll
    for (int c = 0; c < 3; ++c) {
      float ctr = pc[c], nb = pb[c];
      Fr[c]     = (_Float16)(nb - ctr);
      Fr[3 + c] = (_Float16)ctr;
    }
#pragma unroll
    for (int c = 6; c < C2P; ++c) Fr[c] = (_Float16)0.f;
  } else {
    const float* fb = feat + (size_t)b * C * NP;
#pragma unroll 8
    for (int c = 0; c < C; ++c) {
      float ctr = fb[c * NP + n];
      float nb  = fb[c * NP + myIdx];
      Fr[c]     = (_Float16)(nb - ctr);
      Fr[C + c] = (_Float16)ctr;
    }
  }
  __syncthreads();

  constexpr int KS = C2P / 32;
  constexpr int NT = D / 16;
  const _Float16* Fw = &Fsh[w][0][0];

  // ---- pass 1: logits ----
  float accL[2][8];
#pragma unroll
  for (int mt = 0; mt < 2; ++mt)
#pragma unroll
    for (int r = 0; r < 8; ++r) accL[mt][r] = 0.f;

  for (int nt = 0; nt < NT; ++nt) {
    const float bqv = bq[nt * 16 + (l & 15)];
    const float bkv = bk[nt * 16 + (l & 15)];
#pragma unroll
    for (int mt = 0; mt < 2; ++mt) {
      v8f qa = {0.f,0.f,0.f,0.f,0.f,0.f,0.f,0.f};
      v8f ka = {0.f,0.f,0.f,0.f,0.f,0.f,0.f,0.f};
#pragma unroll
      for (int ks = 0; ks < KS; ++ks) {
        v16h a  = load_tile16(Fw,   C2P, mt, ks, l);
        v16h wq = load_tile16(Wq16, C2P, nt, ks, l);
        qa = __builtin_amdgcn_wmma_f32_16x16x32_f16(false, a, false, wq, (short)0, qa, false, false);
        v16h wk = load_tile16(Wk16, C2P, nt, ks, l);
        ka = __builtin_amdgcn_wmma_f32_16x16x32_f16(false, a, false, wk, (short)0, ka, false, false);
      }
#pragma unroll
      for (int r = 0; r < 8; ++r) {
        float p = (qa[r] + bqv) * (ka[r] + bkv);
        p += __shfl_xor(p, 1);
        p += __shfl_xor(p, 2);
        p += __shfl_xor(p, 4);
        p += __shfl_xor(p, 8);
        accL[mt][r] += p;   // row = mt*16 + 8*(l>>4) + r
      }
    }
  }
  if ((l & 15) == 0) {
    const int hh = l >> 4;
#pragma unroll
    for (int mt = 0; mt < 2; ++mt)
#pragma unroll
      for (int r = 0; r < 8; ++r) s_logit[w][mt * 16 + 8 * hh + r] = accL[mt][r];
  }
  __syncthreads();

  // ---- softmax across 32 neighbors (one per lane) ----
  const float rscale = rsqrtf((float)D);
  float lv = s_logit[w][l] * rscale;
  float mx = lv;
#pragma unroll
  for (int m = 1; m < 32; m <<= 1) mx = fmaxf(mx, __shfl_xor(mx, m));
  float e = __expf(lv - mx);
  float sm = e;
#pragma unroll
  for (int m = 1; m < 32; m <<= 1) sm += __shfl_xor(sm, m);
  s_attn[w][l] = e / sm;
  __syncthreads();

  // ---- pass 2: out[d] = sum_k attn[k]*V[k,d] + bv[d] + p[d] ----
  const float* pp = partial + ((size_t)b * NP + n) * 3;
  const float p0 = pp[0], p1 = pp[1], p2 = pp[2];
  const int h = l >> 4;
  for (int nt = 0; nt < NT; ++nt) {
    float o = 0.f;
#pragma unroll
    for (int mt = 0; mt < 2; ++mt) {
      v8f va = {0.f,0.f,0.f,0.f,0.f,0.f,0.f,0.f};
#pragma unroll
      for (int ks = 0; ks < KS; ++ks) {
        v16h a  = load_tile16(Fw,   C2P, mt, ks, l);
        v16h wv = load_tile16(Wv16, C2P, nt, ks, l);
        va = __builtin_amdgcn_wmma_f32_16x16x32_f16(false, a, false, wv, (short)0, va, false, false);
      }
#pragma unroll
      for (int r = 0; r < 8; ++r) o += s_attn[w][mt * 16 + 8 * h + r] * va[r];
    }
    o += __shfl_xor(o, 16);   // merge the two row-halves per column
    if (l < 16) {
      const int d = nt * 16 + l;
      const float pd = Wp[d*3+0]*p0 + Wp[d*3+1]*p1 + Wp[d*3+2]*p2 + bp[d];
      out[((size_t)b * D + d) * NP + n] = o + bv[d] + pd;
    }
  }
}

// ------------------------------------------------------------------
// Channel-wise max + mean over N for concat([x1,x2,x3])
// ------------------------------------------------------------------
__global__ __launch_bounds__(256) void reduce_kernel(const float* __restrict__ x1,
                                                     const float* __restrict__ x2,
                                                     const float* __restrict__ x3,
                                                     float* __restrict__ out) {
  const int b = blockIdx.y;
  const int ch = blockIdx.x;
  const int tid = threadIdx.x;
  const float* src;
  if (ch < 64)       src = x1 + ((size_t)b * 64  + ch)        * NP;
  else if (ch < 128) src = x2 + ((size_t)b * 64  + (ch - 64)) * NP;
  else               src = x3 + ((size_t)b * 128 + (ch - 128)) * NP;
  float mx = -3.4e38f, sm = 0.f;
  for (int i = tid; i < NP; i += 256) { float v = src[i]; mx = fmaxf(mx, v); sm += v; }
  __shared__ float smax[256], ssum[256];
  smax[tid] = mx; ssum[tid] = sm;
  __syncthreads();
  for (int sft = 128; sft > 0; sft >>= 1) {
    if (tid < sft) { smax[tid] = fmaxf(smax[tid], smax[tid + sft]); ssum[tid] += ssum[tid + sft]; }
    __syncthreads();
  }
  if (tid == 0) {
    out[(size_t)b * 512 + ch]       = smax[0];
    out[(size_t)b * 512 + 256 + ch] = ssum[0] / (float)NP;
  }
}

// ------------------------------------------------------------------
extern "C" void kernel_launch(void* const* d_in, const int* in_sizes, int n_in,
                              void* d_out, int out_size, void* d_ws, size_t ws_size,
                              hipStream_t stream) {
  (void)in_sizes; (void)n_in; (void)out_size; (void)ws_size;
  const float* x = (const float*)d_in[0];
  const float *Wq[3], *bq[3], *Wk[3], *bk[3], *Wv[3], *bv[3], *Wp[3], *bp[3];
  for (int li = 0; li < 3; ++li) {
    int base = 1 + li * 8;
    Wq[li] = (const float*)d_in[base + 0]; bq[li] = (const float*)d_in[base + 1];
    Wk[li] = (const float*)d_in[base + 2]; bk[li] = (const float*)d_in[base + 3];
    Wv[li] = (const float*)d_in[base + 4]; bv[li] = (const float*)d_in[base + 5];
    Wp[li] = (const float*)d_in[base + 6]; bp[li] = (const float*)d_in[base + 7];
  }

  char* base8 = (char*)d_ws; size_t off = 0;
  auto alloc = [&](size_t bytes) -> void* {
    void* p = base8 + off; off = (off + bytes + 255) & ~(size_t)255; return p;
  };
  float* partial = (float*)alloc((size_t)B_ * NP * 3 * sizeof(float));
  int*   nbr     = (int*)  alloc((size_t)B_ * NP * KNN_ * sizeof(int));
  float* x1 = (float*)alloc((size_t)B_ * 64  * NP * sizeof(float));
  float* x2 = (float*)alloc((size_t)B_ * 64  * NP * sizeof(float));
  float* x3 = (float*)alloc((size_t)B_ * 128 * NP * sizeof(float));
  const int Dl[3] = {64, 64, 128}, C2l[3] = {6, 128, 128}, C2Pl[3] = {32, 128, 128};
  _Float16* w16[3][3];
  for (int li = 0; li < 3; ++li)
    for (int m = 0; m < 3; ++m)
      w16[li][m] = (_Float16*)alloc((size_t)Dl[li] * C2Pl[li] * sizeof(_Float16));

  fps_kernel<<<B_, 256, 0, stream>>>(x, partial);

  const float* Wsrc[3][3] = {{Wq[0], Wk[0], Wv[0]},
                             {Wq[1], Wk[1], Wv[1]},
                             {Wq[2], Wk[2], Wv[2]}};
  for (int li = 0; li < 3; ++li)
    for (int m = 0; m < 3; ++m) {
      int tot = Dl[li] * C2Pl[li];
      cvt_kernel<<<(tot + 255) / 256, 256, 0, stream>>>(Wsrc[li][m], w16[li][m],
                                                        Dl[li], C2l[li], C2Pl[li]);
    }

  dim3 kgrid(NP / 128, B_);
  dim3 agrid(NP / 4, B_);
  knn_kernel<3, true><<<kgrid, 128, 0, stream>>>(partial, nbr);
  attn_kernel<3, 32, 64, true><<<agrid, 128, 0, stream>>>(partial, nbr, partial,
      w16[0][0], w16[0][1], w16[0][2], bq[0], bk[0], bv[0], Wp[0], bp[0], x1);
  knn_kernel<64, false><<<kgrid, 128, 0, stream>>>(x1, nbr);
  attn_kernel<64, 128, 64, false><<<agrid, 128, 0, stream>>>(x1, nbr, partial,
      w16[1][0], w16[1][1], w16[1][2], bq[1], bk[1], bv[1], Wp[1], bp[1], x2);
  knn_kernel<64, false><<<kgrid, 128, 0, stream>>>(x2, nbr);
  attn_kernel<64, 128, 128, false><<<agrid, 128, 0, stream>>>(x2, nbr, partial,
      w16[2][0], w16[2][1], w16[2][2], bq[2], bk[2], bv[2], Wp[2], bp[2], x3);
  reduce_kernel<<<dim3(256, B_), 256, 0, stream>>>(x1, x2, x3, (float*)d_out);
}
